// PooledAttentionDim2_8538394984705
// MI455X (gfx1250) — matrix operand order repaired and checked by reading, not compiled, not run on hardware
//
#include <hip/hip_runtime.h>
#include <hip/hip_bf16.h>
#include <math.h>
#include <stdint.h>

// Problem constants (match reference setup_inputs)
#define K_HEADS 4
#define D_DIM   64
#define ESTR    72   // padded LDS row stride in halves (64 + 8, kills bank conflicts)

typedef __attribute__((ext_vector_type(16))) _Float16 v16h;
typedef __attribute__((ext_vector_type(2)))  _Float16 h2;
typedef __attribute__((ext_vector_type(8)))  float    v8f;

__device__ __forceinline__ float fast_tanh(float x) {
#if __has_builtin(__builtin_amdgcn_tanhf)
  return __builtin_amdgcn_tanhf(x);
#else
  return tanhf(x);
#endif
}

// A fragment (16x32 f16): source row-major [m][k], row stride ESTR.
// This code's contiguous-k A order (not the ISA table order): lanes 0-15 and 16-31 both hold M=0..15,
// halves differ in K. VGPR v: k0 = half*16 + v*2.
__device__ __forceinline__ v16h load_a_frag(const _Float16* __restrict__ base,
                                            int mrow0, int kb, int lane) {
  const int h  = lane >> 4;
  const int nn = lane & 15;
  const _Float16* rp = base + (mrow0 + nn) * ESTR + kb;
  v16h a;
#pragma unroll
  for (int v = 0; v < 8; ++v) {
    const int k0 = (h << 4) + (v << 1);
    h2 p = *(const h2*)(rp + k0);
    a[2 * v]     = p[0];
    a[2 * v + 1] = p[1];
  }
  return a;
}

// B fragment (32x16 f16) where source is "N-major": src[n][k] so B[k][n]=src[n][k].
// Contiguous-k order, same as load_a_frag: lanes 0-15 hold K=0..15 of column N=lane, lanes 16-31 hold K=16..31.
__device__ __forceinline__ v16h load_b_fragT(const _Float16* __restrict__ base,
                                             int nb, int kb, int lane) {
  const int h  = lane >> 4;
  const int nn = lane & 15;
  const _Float16* rp = base + (nb + nn) * ESTR + kb + (h << 4);
  v16h b;
#pragma unroll
  for (int v = 0; v < 8; ++v) {
    h2 p = *(const h2*)(rp + 2 * v);
    b[2 * v]     = p[0];
    b[2 * v + 1] = p[1];
  }
  return b;
}

// B fragment (32x16 f16) where source is "K-major": src[k][n] directly.
__device__ __forceinline__ v16h load_b_frag(const _Float16* __restrict__ base,
                                            int kb, int nb, int lane) {
  const int h  = lane >> 4;
  const int nn = lane & 15;
  v16h b;
#pragma unroll
  for (int v = 0; v < 8; ++v) {
    b[2 * v]     = base[(kb + (h << 4) + 2 * v)     * ESTR + nb + nn];
    b[2 * v + 1] = base[(kb + (h << 4) + 2 * v + 1) * ESTR + nb + nn];
  }
  return b;
}

__global__ __launch_bounds__(128)
void pooled_attn_kernel(const int* __restrict__ input_,
                        const int* __restrict__ offsets,
                        const float* __restrict__ emb,
                        const float* __restrict__ proj_w,
                        const float* __restrict__ proj_b,
                        const float* __restrict__ att_h,
                        float* __restrict__ out,
                        int nnz, int nseg) {
  __shared__ float    Est[64 * D_DIM];  // async-DMA staging for gathered rows (f32)
  __shared__ _Float16 Eh [64 * ESTR];   // gathered embeddings, f16, rows>=L zeroed
  __shared__ _Float16 Wh [64 * ESTR];   // proj_w  (row a holds proj_w[a][:])
  __shared__ _Float16 Th [64 * ESTR];   // tanh(E*W^T + b)
  __shared__ _Float16 AHh[16 * ESTR];   // att_h transposed: AHh[k][a], rows 4..15 zero
  __shared__ _Float16 WgtH[16 * ESTR];  // softmax weights:  WgtH[k][item], padded zero
  __shared__ float score[64][K_HEADS];
  __shared__ float bias[64];
  __shared__ float mk[K_HEADS], inv_s[K_HEADS];

  const int tid  = threadIdx.x;
  const int lane = tid & 31;
  const int wv   = tid >> 5;   // wave id 0..3
  const int h    = lane >> 4;
  const int nn   = lane & 15;
  const int bseg = blockIdx.x;

  const int off = offsets[bseg];
  const int end = (bseg + 1 < nseg) ? offsets[bseg + 1] : nnz;
  const int Lb  = end - off;   // 50 in this problem; stay general up to 64

  const int c0 = (tid & 15) * 4;   // 16 threads span one 64-wide row (16B chunks)
  const int r0 = tid >> 4;         // 8 rows in flight across the block

  // -------- stage 0a: kick off async DMA gather (global -> LDS, no VGPR) ---
  // Each lane moves one 16B chunk of an embedding row straight into LDS.
  // Tracked by ASYNCcnt; overlaps with the proj_w / att_h conversions below.
#pragma unroll
  for (int i = r0; i < 64; i += 8) {
    if (i < Lb) {
      const size_t row = (size_t)input_[off + i];
      const float* gp  = emb + row * D_DIM + c0;
      const uint32_t lp = (uint32_t)(uintptr_t)(&Est[i * D_DIM + c0]);
      asm volatile("global_load_async_to_lds_b128 %0, %1, off"
                   :: "v"(lp), "v"(gp)
                   : "memory");
    }
  }

  // -------- stage 0b: cooperative LDS fill (overlaps async gather) ---------
#pragma unroll
  for (int r = r0; r < 64; r += 8) {
    const float4 v = *(const float4*)(proj_w + r * 64 + c0);
    Wh[r * ESTR + c0 + 0] = (_Float16)v.x;
    Wh[r * ESTR + c0 + 1] = (_Float16)v.y;
    Wh[r * ESTR + c0 + 2] = (_Float16)v.z;
    Wh[r * ESTR + c0 + 3] = (_Float16)v.w;
  }
  for (int e = tid; e < 16 * 64; e += 128) {
    const int kk = e >> 6, ac = e & 63;
    AHh[kk * ESTR + ac] =
        (kk < K_HEADS) ? (_Float16)att_h[ac * K_HEADS + kk] : (_Float16)0.0f;
  }
  if (tid < 64) bias[tid] = proj_b[tid];

  // -------- stage 0c: wait for DMA, convert gathered rows to f16 -----------
  asm volatile("s_wait_asynccnt 0x0" ::: "memory");
#pragma unroll
  for (int i = r0; i < 64; i += 8) {
    float4 v = make_float4(0.f, 0.f, 0.f, 0.f);
    if (i < Lb) v = *(const float4*)(&Est[i * D_DIM + c0]);
    Eh[i * ESTR + c0 + 0] = (_Float16)v.x;
    Eh[i * ESTR + c0 + 1] = (_Float16)v.y;
    Eh[i * ESTR + c0 + 2] = (_Float16)v.z;
    Eh[i * ESTR + c0 + 3] = (_Float16)v.w;
  }
  __syncthreads();

  // -------- GEMM1: P = E * W^T   (wave wv owns M-tile wv, all 4 N-tiles) ---
  v8f accp[4];
#pragma unroll
  for (int nt = 0; nt < 4; ++nt) accp[nt] = (v8f){};
#pragma unroll
  for (int ks = 0; ks < 2; ++ks) {
    const v16h a = load_a_frag(Eh, 16 * wv, 32 * ks, lane);
#pragma unroll
    for (int nt = 0; nt < 4; ++nt) {
      const v16h bf = load_b_fragT(Wh, 16 * nt, 32 * ks, lane);
      accp[nt] = __builtin_amdgcn_wmma_f32_16x16x32_f16(
          false, a, false, bf, (short)0, accp[nt], false, false);
    }
  }
  // epilogue: bias + tanh -> Th (f16)
#pragma unroll
  for (int nt = 0; nt < 4; ++nt) {
#pragma unroll
    for (int r = 0; r < 8; ++r) {
      const int m  = 16 * wv + (h << 3) + r;
      const int ac = 16 * nt + nn;
      Th[m * ESTR + ac] = (_Float16)fast_tanh(accp[nt][r] + bias[ac]);
    }
  }
  __syncthreads();

  // -------- GEMM2: scores = Th * att_h  (N padded 4 -> 16) -----------------
  v8f accs = (v8f){};
#pragma unroll
  for (int ks = 0; ks < 2; ++ks) {
    const v16h a  = load_a_frag(Th, 16 * wv, 32 * ks, lane);
    const v16h bf = load_b_fragT(AHh, 0, 32 * ks, lane);
    accs = __builtin_amdgcn_wmma_f32_16x16x32_f16(
        false, a, false, bf, (short)0, accs, false, false);
  }
  if (nn < K_HEADS) {
#pragma unroll
    for (int r = 0; r < 8; ++r)
      score[16 * wv + (h << 3) + r][nn] = accs[r];
  }
  __syncthreads();

  // -------- per-bag softmax over L items, each of the K columns ------------
  if (tid < K_HEADS) {
    float m = -3.402823466e38f;
    for (int i = 0; i < Lb; ++i) m = fmaxf(m, score[i][tid]);
    float s = 0.0f;
    for (int i = 0; i < Lb; ++i) s += __expf(score[i][tid] - m);
    mk[tid]    = m;
    inv_s[tid] = 1.0f / s;
  }
  __syncthreads();
  for (int e = tid; e < 16 * 64; e += 128) {
    const int kk = e >> 6, it = e & 63;
    float w = 0.0f;
    if (kk < K_HEADS && it < Lb)
      w = __expf(score[it][kk] - mk[kk]) * inv_s[kk];
    WgtH[kk * ESTR + it] = (_Float16)w;
  }
  __syncthreads();

  // -------- GEMM3: out = Wgt * E  (M padded 4 -> 16; waves split D) --------
  v8f acco = (v8f){};
#pragma unroll
  for (int ks = 0; ks < 2; ++ks) {
    const v16h a  = load_a_frag(WgtH, 0, 32 * ks, lane);
    const v16h bf = load_b_frag(Eh, 32 * ks, 16 * wv, lane);
    acco = __builtin_amdgcn_wmma_f32_16x16x32_f16(
        false, a, false, bf, (short)0, acco, false, false);
  }
  if (h == 0) {
#pragma unroll
    for (int r = 0; r < K_HEADS; ++r)
      out[((size_t)bseg * K_HEADS + r) * D_DIM + 16 * wv + nn] = acco[r];
  }
}

extern "C" void kernel_launch(void* const* d_in, const int* in_sizes, int n_in,
                              void* d_out, int out_size, void* d_ws, size_t ws_size,
                              hipStream_t stream) {
  (void)n_in; (void)out_size; (void)d_ws; (void)ws_size;
  const int*   input_  = (const int*)d_in[0];
  const int*   offsets = (const int*)d_in[1];
  const float* emb     = (const float*)d_in[2];
  const float* proj_w  = (const float*)d_in[3];
  const float* proj_b  = (const float*)d_in[4];
  const float* att_h   = (const float*)d_in[5];
  float*       out     = (float*)d_out;
  const int nnz  = in_sizes[0];
  const int nseg = in_sizes[1];

  pooled_attn_kernel<<<dim3(nseg), dim3(128), 0, stream>>>(
      input_, offsets, emb, proj_w, proj_b, att_h, out, nnz, nseg);
}
